// gated_attention_27599459844515
// MI455X (gfx1250) — matrix operand order repaired and checked by reading, not compiled, not run on hardware
//
#include <hip/hip_runtime.h>
#include <hip/hip_bf16.h>

typedef __attribute__((ext_vector_type(16))) __bf16 v16bf;
typedef __attribute__((ext_vector_type(8)))  __bf16 v8bf;
typedef __attribute__((ext_vector_type(8)))  float  v8f;

constexpr int Bb = 4, Nn = 512, Mm = 512, Dd = 256;

__device__ __forceinline__ __bf16 f2bf(float f) {
  unsigned u = __builtin_bit_cast(unsigned, f);
  unsigned r = (u + 0x7FFFu + ((u >> 16) & 1u)) >> 16;
  unsigned short s = (unsigned short)r;
  return __builtin_bit_cast(__bf16, s);
}

__device__ __forceinline__ v16bf combine16(v8bf lo, v8bf hi) {
  return __builtin_shufflevector(lo, hi, 0,1,2,3,4,5,6,7,8,9,10,11,12,13,14,15);
}

// ------------------- f32 -> bf16 staging -------------------
__global__ void k_cvt_bf16(const float* __restrict__ in, __bf16* __restrict__ out, int n) {
  for (int i = blockIdx.x * blockDim.x + threadIdx.x; i < n; i += gridDim.x * blockDim.x)
    out[i] = f2bf(in[i]);
}

// ------------------- projection GEMM -------------------
// Computes T = A @ WT^T  (A: rows x 256 bf16, WT: 256x256 bf16 row-major, which
// is exactly B^T row-major for the einsum "bnd,ed->bne").
// mode 0: out_scalar[row] = sum_e Wr[e] * tanh(T[row,e] + bias[e])    (additive score)
// mode 2: out_vt[(b*D + e)*M + m] = bf16((T[row,e] + bias[e]) * scale)  (v, transposed)
__global__ void k_proj(const __bf16* __restrict__ A, const __bf16* __restrict__ WT,
                       const float* __restrict__ bias, const float* __restrict__ Wr,
                       float* __restrict__ out_scalar, __bf16* __restrict__ out_vt,
                       int mode, float scale) {
  __shared__ float lds_red[16];
  const int tid  = threadIdx.x;
  const int lane = tid & 31;
  const int wave = tid >> 5;         // 8 waves of 32 (wave32)
  const int row0 = blockIdx.x * 16;  // 16 output rows per block
  const int mrow = lane & 15;
  const int g    = lane >> 4;        // lane half-group
  if (tid < 16) lds_red[tid] = 0.0f;
  __syncthreads();

  const int n0 = wave * 32;          // this wave's 32 output columns (2 tiles)
  v8f acc0 = {}; v8f acc1 = {};
  const __bf16* arow = A + (size_t)(row0 + mrow) * Dd + g * 16;
  for (int k0 = 0; k0 < Dd; k0 += 32) {
    v16bf af = combine16(*(const v8bf*)(arow + k0), *(const v8bf*)(arow + k0 + 8));
    const __bf16* bp0 = WT + (size_t)(n0 + mrow) * Dd + k0 + g * 16;
    const __bf16* bp1 = bp0 + (size_t)16 * Dd;
    v16bf bf0 = combine16(*(const v8bf*)bp0, *(const v8bf*)(bp0 + 8));
    v16bf bf1 = combine16(*(const v8bf*)bp1, *(const v8bf*)(bp1 + 8));
    acc0 = __builtin_amdgcn_wmma_f32_16x16x32_bf16(false, af, false, bf0, (short)0, acc0, false, false);
    acc1 = __builtin_amdgcn_wmma_f32_16x16x32_bf16(false, af, false, bf1, (short)0, acc1, false, false);
  }

  const int c0 = n0 + mrow;
  const int c1 = c0 + 16;
  const float bias0 = bias ? bias[c0] : 0.0f;
  const float bias1 = bias ? bias[c1] : 0.0f;

  if (mode == 2) {
    // acc element r holds (row = row0 + r + 8g, col) -> 8 consecutive m, pack & store
    const int gm0 = row0 + 8 * g;
    const int bb  = gm0 / Mm;
    const int m0  = gm0 - bb * Mm;
    v8bf p0, p1;
    #pragma unroll
    for (int r = 0; r < 8; ++r) {
      p0[r] = f2bf((acc0[r] + bias0) * scale);
      p1[r] = f2bf((acc1[r] + bias1) * scale);
    }
    *(v8bf*)(out_vt + ((size_t)bb * Dd + c0) * Mm + m0) = p0;
    *(v8bf*)(out_vt + ((size_t)bb * Dd + c1) * Mm + m0) = p1;
  } else {
    const float wr0 = Wr[c0], wr1 = Wr[c1];
    #pragma unroll
    for (int r = 0; r < 8; ++r) {
      float s = tanhf(acc0[r] + bias0) * wr0 + tanhf(acc1[r] + bias1) * wr1;
      // reduce over the 16 lanes of this half-group (xor masks < 16 stay in-half)
      s += __shfl_xor(s, 1); s += __shfl_xor(s, 2);
      s += __shfl_xor(s, 4); s += __shfl_xor(s, 8);
      if (mrow == 0) atomicAdd(&lds_red[r + 8 * g], s);
    }
    __syncthreads();
    if (tid < 16) out_scalar[row0 + tid] = lds_red[tid];
  }
}

// ------------------- fused attention GEMM + residual + LayerNorm -------------------
__global__ void k_attn_ln(const float* __restrict__ a, const float* __restrict__ bsc,
                          const float* __restrict__ mask, const __bf16* __restrict__ v_t,
                          const float* __restrict__ x, const float* __restrict__ gamma,
                          const float* __restrict__ beta, float* __restrict__ out) {
  __shared__ __bf16 lds_w[16 * 520];   // 16 rows x 512 scores, stride 520 to spread banks
  __shared__ float lds_sum[16];
  __shared__ float lds_sq[16];
  const int tid  = threadIdx.x;
  const int lane = tid & 31;
  const int wave = tid >> 5;
  const int b    = blockIdx.x >> 5;           // 32 row-tiles per batch (N/16)
  const int n0   = (blockIdx.x & 31) * 16;

  if (tid < 16) { lds_sum[tid] = 0.0f; lds_sq[tid] = 0.0f; }

  // Phase 1: build score stripe w[16 x 512] cooperatively into LDS (consumed once).
  const float* ap = a + b * Nn + n0;
  const float* bp = bsc + b * Mm;
  const float* mp = mask + ((size_t)b * Nn + n0) * Mm;
  for (int idx = tid; idx < 16 * Mm; idx += 256) {
    int row = idx >> 9;          // Mm == 512
    int m   = idx & (Mm - 1);
    float s = tanhf(ap[row] + bp[m]);
    s = fmaxf(s, 0.0f) * mp[(size_t)row * Mm + m];
    lds_w[row * 520 + m] = f2bf(s);
  }
  __syncthreads();

  // Phase 2: add = w @ v via bf16 WMMA; B fragments read from transposed v.
  const int mrow = lane & 15;
  const int g    = lane >> 4;
  const int d0   = wave * 32;
  v8f acc0 = {}; v8f acc1 = {};
  const __bf16* abase  = &lds_w[mrow * 520 + g * 16];
  const __bf16* bbase0 = v_t + ((size_t)b * Dd + d0 + mrow) * Mm + g * 16;
  const __bf16* bbase1 = bbase0 + (size_t)16 * Mm;
  for (int k0 = 0; k0 < Mm; k0 += 32) {
    v16bf af  = combine16(*(const v8bf*)(abase + k0), *(const v8bf*)(abase + k0 + 8));
    v16bf bf0 = combine16(*(const v8bf*)(bbase0 + k0), *(const v8bf*)(bbase0 + k0 + 8));
    v16bf bf1 = combine16(*(const v8bf*)(bbase1 + k0), *(const v8bf*)(bbase1 + k0 + 8));
    __builtin_prefetch(bbase0 + k0 + 64, 0, 1);   // global_prefetch_b8
    acc0 = __builtin_amdgcn_wmma_f32_16x16x32_bf16(false, af, false, bf0, (short)0, acc0, false, false);
    acc1 = __builtin_amdgcn_wmma_f32_16x16x32_bf16(false, af, false, bf1, (short)0, acc1, false, false);
  }

  // Phase 3: residual + LayerNorm statistics (per-row mean / mean-square).
  const int c0 = d0 + mrow, c1 = c0 + 16;
  const float* xp = x + ((size_t)b * Nn + n0) * Dd;
  float h0[8], h1[8];
  #pragma unroll
  for (int r = 0; r < 8; ++r) {
    int row = r + 8 * g;
    h0[r] = xp[(size_t)row * Dd + c0] + acc0[r];
    h1[r] = xp[(size_t)row * Dd + c1] + acc1[r];
    float s = h0[r] + h1[r];
    float q = h0[r] * h0[r] + h1[r] * h1[r];
    s += __shfl_xor(s, 1); s += __shfl_xor(s, 2); s += __shfl_xor(s, 4); s += __shfl_xor(s, 8);
    q += __shfl_xor(q, 1); q += __shfl_xor(q, 2); q += __shfl_xor(q, 4); q += __shfl_xor(q, 8);
    if (mrow == 0) { atomicAdd(&lds_sum[row], s); atomicAdd(&lds_sq[row], q); }
  }
  __syncthreads();

  const float ga0 = gamma[c0], ga1 = gamma[c1];
  const float be0 = beta[c0],  be1 = beta[c1];
  float* op = out + ((size_t)b * Nn + n0) * Dd;
  #pragma unroll
  for (int r = 0; r < 8; ++r) {
    int row = r + 8 * g;
    float mu  = lds_sum[row] * (1.0f / Dd);
    float var = lds_sq[row] * (1.0f / Dd) - mu * mu;
    float inv = rsqrtf(var + 1e-5f);
    op[(size_t)row * Dd + c0] = (h0[r] - mu) * inv * ga0 + be0;
    op[(size_t)row * Dd + c1] = (h1[r] - mu) * inv * ga1 + be1;
  }
}

extern "C" void kernel_launch(void* const* d_in, const int* in_sizes, int n_in,
                              void* d_out, int out_size, void* d_ws, size_t ws_size,
                              hipStream_t stream) {
  const float* x     = (const float*)d_in[0];
  const float* c     = (const float*)d_in[1];
  const float* mask  = (const float*)d_in[2];
  const float* Wq    = (const float*)d_in[3];
  const float* bq    = (const float*)d_in[4];
  const float* Wk    = (const float*)d_in[5];
  const float* Wr    = (const float*)d_in[6];
  const float* Wv    = (const float*)d_in[7];
  const float* bv    = (const float*)d_in[8];
  const float* gamma = (const float*)d_in[9];
  const float* beta  = (const float*)d_in[10];
  float* out = (float*)d_out;

  char* ws = (char*)d_ws;
  size_t off = 0;
  auto alloc = [&](size_t bytes) {
    void* p = ws + off;
    off = (off + bytes + 255) & ~(size_t)255;
    return p;
  };
  __bf16* xb  = (__bf16*)alloc((size_t)Bb * Nn * Dd * 2);
  __bf16* cb  = (__bf16*)alloc((size_t)Bb * Mm * Dd * 2);
  __bf16* Wqb = (__bf16*)alloc((size_t)Dd * Dd * 2);
  __bf16* Wkb = (__bf16*)alloc((size_t)Dd * Dd * 2);
  __bf16* Wvb = (__bf16*)alloc((size_t)Dd * Dd * 2);
  float*  aS  = (float*)alloc((size_t)Bb * Nn * 4);
  float*  sS  = (float*)alloc((size_t)Bb * Mm * 4);
  __bf16* vT  = (__bf16*)alloc((size_t)Bb * Dd * Mm * 2);

  k_cvt_bf16<<<256, 256, 0, stream>>>(x, xb, Bb * Nn * Dd);
  k_cvt_bf16<<<256, 256, 0, stream>>>(c, cb, Bb * Mm * Dd);
  k_cvt_bf16<<<64, 256, 0, stream>>>(Wq, Wqb, Dd * Dd);
  k_cvt_bf16<<<64, 256, 0, stream>>>(Wk, Wkb, Dd * Dd);
  k_cvt_bf16<<<64, 256, 0, stream>>>(Wv, Wvb, Dd * Dd);

  // a[b,n] = Wr . tanh(x Wq^T + bq);  s[b,m] = Wr . tanh(c Wk^T);  vT = ((c Wv^T + bv)/sqrt(D))^T
  k_proj<<<(Bb * Nn) / 16, 256, 0, stream>>>(xb, Wqb, bq, Wr, aS, nullptr, 0, 1.0f);
  k_proj<<<(Bb * Mm) / 16, 256, 0, stream>>>(cb, Wkb, nullptr, Wr, sS, nullptr, 0, 1.0f);
  k_proj<<<(Bb * Mm) / 16, 256, 0, stream>>>(cb, Wvb, bv, nullptr, nullptr, vT, 2, 0.0625f);

  k_attn_ln<<<Bb * (Nn / 16), 256, 0, stream>>>(aS, sS, mask, vT, x, gamma, beta, out);
}
